// PairedLookup_16509854286522
// MI455X (gfx1250) — matrix operand order repaired and checked by reading, not compiled, not run on hardware
//
#include <hip/hip_runtime.h>

typedef float  v2f   __attribute__((ext_vector_type(2)));
typedef float  v8f   __attribute__((ext_vector_type(8)));
typedef __bf16 v8bf  __attribute__((ext_vector_type(8)));
typedef __bf16 v16bf __attribute__((ext_vector_type(16)));

#define DD   256
#define NTAB 4096
#define BQ   1024

// ---------------- Kernel 0: reset argmin keys (must run every call) ----------
__global__ void pl_init_keys(unsigned long long* __restrict__ keys) {
    int b = blockIdx.x * blockDim.x + threadIdx.x;
    if (b < BQ) keys[b] = 0xFFFFFFFFFFFFFFFFULL;
}

// ---------------- Kernel 1: Y = X @ W^T (exact, f32 WMMA) --------------------
// One wave per 16x16 tile. A-frag (16x4 f32): lanes 0-15 K={0,1}, 16-31 K={2,3}.
// B-frag (4x16 f32): VGPR0 = K-rows {0,2}, VGPR1 = {1,3}; both are float2 loads.
__global__ void pl_gemm_xwT(const float* __restrict__ X,
                            const float* __restrict__ W,
                            float* __restrict__ Y) {
    const int m0   = blockIdx.x * 16;
    const int n0   = blockIdx.y * 16;
    const int lane = threadIdx.x;        // single wave
    const int l    = lane & 15;
    const int half = lane >> 4;

    const float* xrow = X + (m0 + l) * DD + 2 * half;
    const float* wrow = W + (n0 + l) * DD + 2 * half;

    v8f c = {};
#pragma unroll 8
    for (int k = 0; k < DD; k += 4) {
        v2f a = *(const v2f*)(xrow + k);
        v2f b = *(const v2f*)(wrow + k);
        c = __builtin_amdgcn_wmma_f32_16x16x4_f32(false, a, false, b,
                                                  (short)0, c, false, false);
    }
    float* out = Y + (m0 + half * 8) * DD + (n0 + l);
#pragma unroll
    for (int r = 0; r < 8; ++r) out[r * DD] = c[r];
}

// ---------------- Kernel 2: f32 -> bf16 (RNE), 8 elems/thread ----------------
__global__ void pl_cvt_bf16(const float* __restrict__ src,
                            __bf16* __restrict__ dst, int n) {
    int i = (blockIdx.x * blockDim.x + threadIdx.x) * 8;
    if (i >= n) return;
    float4 f0 = *(const float4*)(src + i);
    float4 f1 = *(const float4*)(src + i + 4);
    v8bf o;
    o[0] = (__bf16)f0.x; o[1] = (__bf16)f0.y; o[2] = (__bf16)f0.z; o[3] = (__bf16)f0.w;
    o[4] = (__bf16)f1.x; o[5] = (__bf16)f1.y; o[6] = (__bf16)f1.z; o[7] = (__bf16)f1.w;
    *(uint4*)(dst + i) = *(uint4*)&o;
}

// ---------------- Kernel 3: row norms of As (f32) ----------------------------
__global__ void pl_row_norms(const float* __restrict__ A, float* __restrict__ nrm) {
    __shared__ float red[256];
    const int row = blockIdx.x;
    float v = A[row * DD + threadIdx.x];
    red[threadIdx.x] = v * v;
    __syncthreads();
    for (int s = 128; s > 0; s >>= 1) {
        if (threadIdx.x < s) red[threadIdx.x] += red[threadIdx.x + s];
        __syncthreads();
    }
    if (threadIdx.x == 0) nrm[row] = red[0];
}

// ---------------- Kernel 4: bf16 WMMA score + argmin-distance ----------------
// Block = 256 threads = 8 waves. Block stages 16 bf16 Y rows in LDS (8 KB).
// Each wave computes a 16(M) x 32(N) tile with two accumulators sharing the
// A-fragment; K-loop = 8 steps of v_wmma_f32_16x16x32_bf16 (x2 accumulators).
// dist(m, n) = ||a_n||^2 - 2 * (y_m . a_n)   (||y||^2 constant -> dropped)
__global__ void pl_score_argmin(const __bf16* __restrict__ Yb,
                                const __bf16* __restrict__ Ab,
                                const float* __restrict__ nrmA,
                                unsigned long long* __restrict__ keys) {
    __shared__ __align__(16) __bf16 ytile[16 * DD];   // 8 KB
    const int m0  = blockIdx.x * 16;
    const int tid = threadIdx.x;                      // 0..255

    // cooperative stage: 8192 bytes = 512 uint4, 2 per thread
    {
        const uint4* src = (const uint4*)(Yb + (size_t)m0 * DD);
        uint4*       dst = (uint4*)ytile;
        dst[tid]       = src[tid];
        dst[tid + 256] = src[tid + 256];
    }
    __syncthreads();

    const int wave = tid >> 5;
    const int lane = tid & 31;
    const int l    = lane & 15;
    const int half = lane >> 4;
    const int n0   = (blockIdx.y * 8 + wave) * 32;    // 32 columns per wave

    // A-frag (16x32 bf16): lane row = m0+l; contiguous K chunk at 16*half
    const __bf16* yrow = ytile + l * DD;
    // B-frag (32x16 bf16): column = nb+l, contiguous K chunk at 16*half
    const __bf16* b0row = Ab + (size_t)(n0 + l)      * DD + 16 * half;
    const __bf16* b1row = Ab + (size_t)(n0 + 16 + l) * DD + 16 * half;

    v8f c0 = {}, c1 = {};
#pragma unroll
    for (int k = 0; k < DD; k += 32) {
        v16bf a, b0, b1;
        ((uint4*)&a)[0]  = *(const uint4*)(yrow + k + 16 * half);
        ((uint4*)&a)[1]  = *(const uint4*)(yrow + k + 16 * half + 8);
        ((uint4*)&b0)[0] = *(const uint4*)(b0row + k);
        ((uint4*)&b0)[1] = *(const uint4*)(b0row + k + 8);
        ((uint4*)&b1)[0] = *(const uint4*)(b1row + k);
        ((uint4*)&b1)[1] = *(const uint4*)(b1row + k + 8);
        c0 = __builtin_amdgcn_wmma_f32_16x16x32_bf16(false, a, false, b0,
                                                     (short)0, c0, false, false);
        c1 = __builtin_amdgcn_wmma_f32_16x16x32_bf16(false, a, false, b1,
                                                     (short)0, c1, false, false);
    }

    const float nn0 = nrmA[n0 + l];
    const float nn1 = nrmA[n0 + 16 + l];
    const int   mbase = m0 + half * 8;
#pragma unroll
    for (int r = 0; r < 8; ++r) {
        float d0 = nn0 - 2.0f * c0[r];
        float d1 = nn1 - 2.0f * c1[r];
        unsigned int k0 = __float_as_uint(d0);
        unsigned int k1 = __float_as_uint(d1);
        k0 ^= (k0 & 0x80000000u) ? 0xFFFFFFFFu : 0x80000000u;  // order-preserving
        k1 ^= (k1 & 0x80000000u) ? 0xFFFFFFFFu : 0x80000000u;
        unsigned long long key0 = ((unsigned long long)k0 << 32) | (unsigned int)(n0 + l);
        unsigned long long key1 = ((unsigned long long)k1 << 32) | (unsigned int)(n0 + 16 + l);
        atomicMin(keys + (mbase + r), key0);   // ties -> lowest index ("first match")
        atomicMin(keys + (mbase + r), key1);
    }
}

// ---------------- Kernel 5: out[b] = Y[b] . Bs[idx_b] (f32) ------------------
__global__ void pl_gather_dot(const float* __restrict__ Y,
                              const float* __restrict__ Bs,
                              const unsigned long long* __restrict__ keys,
                              float* __restrict__ out) {
    __shared__ float red[256];
    const int b = blockIdx.x;
    const unsigned int idx = (unsigned int)(keys[b] & 0xFFFFFFFFu);
    red[threadIdx.x] = Y[b * DD + threadIdx.x] * Bs[(size_t)idx * DD + threadIdx.x];
    __syncthreads();
    for (int s = 128; s > 0; s >>= 1) {
        if (threadIdx.x < s) red[threadIdx.x] += red[threadIdx.x + s];
        __syncthreads();
    }
    if (threadIdx.x == 0) out[b] = red[0];
}

// ---------------- Launch -----------------------------------------------------
extern "C" void kernel_launch(void* const* d_in, const int* in_sizes, int n_in,
                              void* d_out, int out_size, void* d_ws, size_t ws_size,
                              hipStream_t stream) {
    const float* batch  = (const float*)d_in[0];  // (B, D)
    const float* weight = (const float*)d_in[1];  // (D, D)
    const float* As     = (const float*)d_in[2];  // (N, D)
    const float* Bs     = (const float*)d_in[3];  // (N, D)
    float*       out    = (float*)d_out;          // (B, 1)

    char* ws = (char*)d_ws;
    size_t off = 0;
    float*   Y    = (float*)(ws + off);   off += (size_t)BQ * DD * sizeof(float);    // 1 MB
    __bf16*  Yb   = (__bf16*)(ws + off);  off += (size_t)BQ * DD * sizeof(__bf16);   // 512 KB
    __bf16*  Ab   = (__bf16*)(ws + off);  off += (size_t)NTAB * DD * sizeof(__bf16); // 2 MB
    float*   nrmA = (float*)(ws + off);   off += (size_t)NTAB * sizeof(float);       // 16 KB
    unsigned long long* keys = (unsigned long long*)(ws + off);                      // 8 KB

    pl_init_keys   <<<(BQ + 255) / 256, 256, 0, stream>>>(keys);
    pl_gemm_xwT    <<<dim3(BQ / 16, DD / 16), 32, 0, stream>>>(batch, weight, Y);
    pl_row_norms   <<<NTAB, 256, 0, stream>>>(As, nrmA);
    pl_cvt_bf16    <<<(BQ * DD / 8 + 255) / 256, 256, 0, stream>>>(Y, Yb, BQ * DD);
    pl_cvt_bf16    <<<(NTAB * DD / 8 + 255) / 256, 256, 0, stream>>>(As, Ab, NTAB * DD);
    pl_score_argmin<<<dim3(BQ / 16, NTAB / 256), 256, 0, stream>>>(Yb, Ab, nrmA, keys);
    pl_gather_dot  <<<BQ, 256, 0, stream>>>(Y, Bs, keys, out);
}
